// LSTM_FF_FWPModel_70153995813283
// MI455X (gfx1250) — matrix operand order repaired and checked by reading, not compiled, not run on hardware
//
#include <hip/hip_runtime.h>
#include <hip/hip_bf16.h>
#include <math.h>
#include <stdint.h>

// ---------------------------------------------------------------------------
// Model constants (from the reference)
// ---------------------------------------------------------------------------
#define T_STEPS   1024
#define NUM_INS   256
#define HID       256
#define G4        1024          // 4*HID
#define UU        340           // ceil(sqrt(P))
#define SLOW_OUTS 1362          // 4*U + 2
#define SLOW_PAD  1376          // padded to multiple of 16 for WMMA tiles
#define P_TOT     115456
#define P_STAGE_A 49408         // fast layers 0-1 (W0+b0+W1+b1) -> block 0 LDS
#define P_STAGE_B (P_TOT - P_STAGE_A)  // 66048, fast layers 2-4 -> block 1 LDS
#define KDIM      256           // every GEMM in this model has K == 256

typedef __attribute__((ext_vector_type(16))) __bf16 v16bf;
typedef __attribute__((ext_vector_type(8)))  __bf16 v8bf;
typedef __attribute__((ext_vector_type(8)))  float  v8f;

__device__ __forceinline__ float sigmoidf_(float x) { return 1.0f / (1.0f + expf(-x)); }

// ---------------------------------------------------------------------------
// gfx1250 async global->LDS copy path (ASYNCcnt-tracked), with fallback.
// Builtin signature (from compiler diagnostic): param 0 is AS(1) int*,
// param 1 is the LDS destination (AS(3) int*), then imm offset + cpol.
// ---------------------------------------------------------------------------
#if defined(__AMDGCN__) && __has_builtin(__builtin_amdgcn_global_load_async_to_lds_b32)
#define HAVE_ASYNC_LDS 1
#else
#define HAVE_ASYNC_LDS 0
#endif

#if HAVE_ASYNC_LDS
typedef __attribute__((address_space(1))) int as1_int;
typedef __attribute__((address_space(3))) int as3_int;
__device__ __forceinline__ void async_ld_b32(const float* g, float* l) {
    // per-lane: LDS[l] = MEM[g]; tracked by ASYNCcnt
    __builtin_amdgcn_global_load_async_to_lds_b32((as1_int*)g, (as3_int*)l, 0, 0);
}
__device__ __forceinline__ void wait_async0() {
#if __has_builtin(__builtin_amdgcn_s_wait_asynccnt)
    __builtin_amdgcn_s_wait_asynccnt(0);
#else
    asm volatile("s_wait_asynccnt 0x0" ::: "memory");
#endif
}
#endif

// ---------------------------------------------------------------------------
// Elementwise helpers
// ---------------------------------------------------------------------------
__global__ void k_f32_to_bf16(const float* __restrict__ in, __bf16* __restrict__ out, int n) {
    int i = blockIdx.x * blockDim.x + threadIdx.x;
    if (i < n) out[i] = (__bf16)in[i];
}

__global__ void k_zero_f32(float* __restrict__ p, int n) {
    int i = blockIdx.x * blockDim.x + threadIdx.x;
    if (i < n) p[i] = 0.0f;
}

// W is (K x N) row-major f32; produce WT = B^T as (Npad x K) row-major bf16.
__global__ void k_transpose_bf16(const float* __restrict__ W, __bf16* __restrict__ WT,
                                 int K, int N, int Npad) {
    int i = blockIdx.x * blockDim.x + threadIdx.x;
    if (i >= Npad * K) return;
    int n = i / K;
    int k = i - n * K;
    WT[i] = (n < N) ? (__bf16)W[(size_t)k * N + n] : (__bf16)0.0f;
}

// Pack Whh (K x N f32) into (K/2 x N) u32 words: lo16 = bf16(W[2k][j]),
// hi16 = bf16(W[2k+1][j]).
__global__ void k_pack_bf16x2(const float* __restrict__ in, uint32_t* __restrict__ out,
                              int K, int N) {
    int i = blockIdx.x * blockDim.x + threadIdx.x;
    if (i >= (K / 2) * N) return;
    int k2 = i / N;
    int j  = i - k2 * N;
    __bf16 b0 = (__bf16)in[(size_t)(2 * k2) * N + j];
    __bf16 b1 = (__bf16)in[(size_t)(2 * k2 + 1) * N + j];
    uint32_t u0 = (uint32_t)__builtin_bit_cast(unsigned short, b0);
    uint32_t u1 = (uint32_t)__builtin_bit_cast(unsigned short, b1);
    out[i] = u0 | (u1 << 16);
}

// ---------------------------------------------------------------------------
// bf16 WMMA GEMM:  C(MxN) = A(MxK) @ B(KxN) + bias1 (+ bias2), optional ReLU,
// optional bf16 copy of C. K == 256, fully unrolled. One wave per 16x16 tile.
// ---------------------------------------------------------------------------
__global__ __launch_bounds__(128)
void k_gemm_bf16(const __bf16* __restrict__ A,    // M x 256 row-major
                 const __bf16* __restrict__ BT,   // Npad x 256 row-major (B^T)
                 const float*  __restrict__ bias1,
                 const float*  __restrict__ bias2,
                 float*        __restrict__ C, int ldc,
                 __bf16*       __restrict__ Cbf,  // nullable, stride Nreal
                 int M, int Nreal, int Npad, int act)
{
    const int lane = threadIdx.x & 31;
    const int wid  = threadIdx.x >> 5;
    const int ntn  = Npad >> 4;
    const int tile = blockIdx.x * (blockDim.x >> 5) + wid;
    if (tile >= ntn * (M >> 4)) return;
    const int tn = tile % ntn;
    const int tm = tile / ntn;

    const int mrow = lane & 15;
    const int half = lane >> 4;

    // A fragment (16x32): lane<16 -> K 0..7 & 16..23 ; lane>=16 -> 8..15 & 24..31
    const __bf16* ap = A  + (size_t)(tm * 16 + mrow) * KDIM + half * 8;
    // B fragment (32x16): lane<16 -> K 0..7 & 16..23 ; lane>=16 -> 8..15 & 24..31 (column = mrow)
    const __bf16* bp = BT + (size_t)(tn * 16 + mrow) * KDIM + half * 8;

    v8f acc = {};
#pragma unroll
    for (int kc = 0; kc < KDIM / 32; ++kc) {
        v8bf a0 = *(const v8bf*)(ap + kc * 32);
        v8bf a1 = *(const v8bf*)(ap + kc * 32 + 16);
        v8bf b0 = *(const v8bf*)(bp + kc * 32);
        v8bf b1 = *(const v8bf*)(bp + kc * 32 + 16);
        v16bf av, bv;
#pragma unroll
        for (int i = 0; i < 8; ++i) {
            av[i] = a0[i]; av[i + 8] = a1[i];
            bv[i] = b0[i]; bv[i + 8] = b1[i];
        }
        acc = __builtin_amdgcn_wmma_f32_16x16x32_bf16(
            false, av, false, bv, (short)0, acc, false, false);
    }

    const int n = tn * 16 + (lane & 15);
    if (n < Nreal) {
        float b = bias1 ? bias1[n] : 0.0f;
        if (bias2) b += bias2[n];
#pragma unroll
        for (int i = 0; i < 8; ++i) {
            int m = tm * 16 + half * 8 + i;
            float v = acc[i] + b;
            if (act) v = fmaxf(v, 0.0f);
            C[(size_t)m * ldc + n] = v;
            if (Cbf) Cbf[(size_t)m * Nreal + n] = (__bf16)v;
        }
    }
}

// ---------------------------------------------------------------------------
// LSTM recurrence scan; Whh packed bf16x2 halves the per-step L2 stream.
// ---------------------------------------------------------------------------
__global__ __launch_bounds__(1024)
void k_lstm_scan(const float*    __restrict__ Aproj, // T x 1024
                 const uint32_t* __restrict__ Whp,   // 128 x 1024 packed bf16x2
                 float*          __restrict__ Hout,  // T x 256
                 __bf16*         __restrict__ Hbf)   // T x 256
{
    __shared__ float h[HID];
    __shared__ float c[HID];
    __shared__ float g[G4];
    const int j = threadIdx.x;
    if (j < HID) { h[j] = 0.0f; c[j] = 0.0f; }
    __syncthreads();

    for (int t = 0; t < T_STEPS; ++t) {
        __builtin_prefetch(Aproj + (size_t)(t + 1) * G4 + j, 0, 3);

        float acc = Aproj[(size_t)t * G4 + j];
#pragma unroll 8
        for (int k2 = 0; k2 < HID / 2; ++k2) {
            uint32_t w = Whp[(size_t)k2 * G4 + j];
            float w0 = __uint_as_float(w << 16);           // bf16 lo -> f32
            float w1 = __uint_as_float(w & 0xffff0000u);   // bf16 hi -> f32
            acc += h[2 * k2] * w0 + h[2 * k2 + 1] * w1;
        }
        g[j] = acc;
        __syncthreads();
        if (j < HID) {
            float ig = sigmoidf_(g[j]);
            float fg = sigmoidf_(g[HID + j]);
            float gg = tanhf(g[2 * HID + j]);
            float og = sigmoidf_(g[3 * HID + j]);
            float cn = fg * c[j] + ig * gg;
            float hn = og * tanhf(cn);
            c[j] = cn;
            h[j] = hn;
            Hout[(size_t)t * HID + j] = hn;
            Hbf[(size_t)t * HID + j]  = (__bf16)hn;
        }
        __syncthreads();
    }
}

// ---------------------------------------------------------------------------
// Fast-weight scan: 2-block producer/consumer pipeline, dflat LDS-resident.
// The per-step u-row / x-row staging into LDS uses gfx1250 async global->LDS
// loads (ASYNCcnt) when the toolchain exposes them.
// ---------------------------------------------------------------------------
struct FastParams {
    const float* W[5];
    const float* b[5];
};

__global__ __launch_bounds__(1024)
void k_fast_pipe(const float* __restrict__ u,     // T x SLOW_PAD (first 1362 valid)
                 const float* __restrict__ x,     // T x 256
                 FastParams fp,
                 float* __restrict__ hmid,        // T x 128 handoff
                 int*   __restrict__ flags,       // T ints, pre-zeroed
                 float* __restrict__ out)         // T x 256
{
    extern __shared__ float dfl[];                // stage-local dflat slice
    __shared__ float a1[UU], a2[UU], s1v[UU], s2v[UU];
    __shared__ float hbuf[256], zbuf[256];
    __shared__ float sred[2];
    const int j = threadIdx.x;
    const int stage = blockIdx.x;
    const int myP  = (stage == 0) ? P_STAGE_A : P_STAGE_B;
    const int base = (stage == 0) ? 0 : P_STAGE_A;

    for (int p = j; p < myP; p += 1024) dfl[p] = 0.0f;
    __syncthreads();

    for (int t = 0; t < T_STEPS; ++t) {
        const float* ui = u + (size_t)t * SLOW_PAD;

#if HAVE_ASYNC_LDS
        // Async-stage the shared per-step vectors into LDS (ASYNCcnt path).
        if (j < UU) {
            async_ld_b32(ui + j,          &a1[j]);
            async_ld_b32(ui + UU + j,     &a2[j]);
            async_ld_b32(ui + 2 * UU + j, &s1v[j]);
            async_ld_b32(ui + 3 * UU + j, &s2v[j]);
        }
        if (stage == 0 && j < 256)
            async_ld_b32(x + (size_t)t * 256 + j, &hbuf[j]);
        if (j == 0) {
            sred[0] = sigmoidf_(ui[4 * UU]);
            sred[1] = sigmoidf_(ui[4 * UU + 1]);
        }
        wait_async0();          // drain this wave's ASYNCcnt before the barrier
        __syncthreads();
#else
        if (j < UU) {
            a1[j]  = ui[j];
            a2[j]  = ui[UU + j];
            s1v[j] = ui[2 * UU + j];
            s2v[j] = ui[3 * UU + j];
        }
        if (j == 0) {
            sred[0] = sigmoidf_(ui[4 * UU]);
            sred[1] = sigmoidf_(ui[4 * UU + 1]);
        }
        __syncthreads();
#endif
        const float sa = sred[0], ss = sred[1];

        // dflat_half += outer(a1,a2).ravel()*sa - outer(s1,s2).ravel()*ss
        for (int p = j; p < myP; p += 1024) {
            int pg  = base + p;
            int r   = pg / UU;
            int col = pg - r * UU;
            dfl[p] += a1[r] * a2[col] * sa - s1v[r] * s2v[col] * ss;
        }

        if (stage == 0) {
#if !HAVE_ASYNC_LDS
            if (j < 256) hbuf[j] = x[(size_t)t * 256 + j];
#endif
            __syncthreads();
            // layer 0: 256 -> 128, ReLU. Local: W0 at 0, b0 at 32768.
            if (j < 128) {
                float acc = fp.b[0][j] + dfl[32768 + j];
                for (int i = 0; i < 256; ++i)
                    acc += hbuf[i] * (fp.W[0][i * 128 + j] + dfl[i * 128 + j]);
                zbuf[j] = fmaxf(acc, 0.0f);
            }
            __syncthreads();
            if (j < 128) hbuf[j] = zbuf[j];
            __syncthreads();
            // layer 1: 128 -> 128, ReLU -> hmid. Local: W1 at 32896, b1 at 49280.
            if (j < 128) {
                float acc = fp.b[1][j] + dfl[32896 + 16384 + j];
                for (int i = 0; i < 128; ++i)
                    acc += hbuf[i] * (fp.W[1][i * 128 + j] + dfl[32896 + i * 128 + j]);
                hmid[(size_t)t * 128 + j] = fmaxf(acc, 0.0f);
            }
            __threadfence();
            __syncthreads();
            if (j == 0)
                __hip_atomic_store(&flags[t], 1, __ATOMIC_RELEASE, __HIP_MEMORY_SCOPE_AGENT);
            __syncthreads();
        } else {
            if (j == 0) {
                while (__hip_atomic_load(&flags[t], __ATOMIC_ACQUIRE,
                                         __HIP_MEMORY_SCOPE_AGENT) == 0)
                    __builtin_amdgcn_s_sleep(1);
            }
            __syncthreads();
            if (j < 128) hbuf[j] = hmid[(size_t)t * 128 + j];
            __syncthreads();
            // layer 2: 128 -> 128, ReLU. Local: W2 at 0, b2 at 16384.
            if (j < 128) {
                float acc = fp.b[2][j] + dfl[16384 + j];
                for (int i = 0; i < 128; ++i)
                    acc += hbuf[i] * (fp.W[2][i * 128 + j] + dfl[i * 128 + j]);
                zbuf[j] = fmaxf(acc, 0.0f);
            }
            __syncthreads();
            if (j < 128) hbuf[j] = zbuf[j];
            __syncthreads();
            // layer 3: 128 -> 128, ReLU. Local: W3 at 16512, b3 at 32896.
            if (j < 128) {
                float acc = fp.b[3][j] + dfl[16512 + 16384 + j];
                for (int i = 0; i < 128; ++i)
                    acc += hbuf[i] * (fp.W[3][i * 128 + j] + dfl[16512 + i * 128 + j]);
                zbuf[j] = fmaxf(acc, 0.0f);
            }
            __syncthreads();
            if (j < 128) hbuf[j] = zbuf[j];
            __syncthreads();
            // layer 4: 128 -> 256, softmax. Local: W4 at 33024, b4 at 65792.
            if (j < 256) {
                float acc = fp.b[4][j] + dfl[33024 + 32768 + j];
                for (int i = 0; i < 128; ++i)
                    acc += hbuf[i] * (fp.W[4][i * 256 + j] + dfl[33024 + i * 256 + j]);
                zbuf[j] = acc;
            }
            __syncthreads();
            if (j == 0) {
                float mx = zbuf[0];
                for (int i = 1; i < 256; ++i) mx = fmaxf(mx, zbuf[i]);
                sred[0] = mx;
            }
            __syncthreads();
            if (j < 256) zbuf[j] = expf(zbuf[j] - sred[0]);
            __syncthreads();
            if (j == 0) {
                float sm = 0.0f;
                for (int i = 0; i < 256; ++i) sm += zbuf[i];
                sred[1] = sm;
            }
            __syncthreads();
            if (j < 256) out[(size_t)t * 256 + j] = zbuf[j] / sred[1];
            __syncthreads();
        }
    }
}

// ---------------------------------------------------------------------------
// Host launcher
// ---------------------------------------------------------------------------
extern "C" void kernel_launch(void* const* d_in, const int* in_sizes, int n_in,
                              void* d_out, int out_size, void* d_ws, size_t ws_size,
                              hipStream_t stream) {
    if (n_in < 23) return;

    const float *x, *in_W, *in_b, *out_W, *out_b;
    const float *Wih[2], *Whh[2], *bih[2], *bhh[2];
    const float *fW[5], *fb[5];

    // setup_inputs order may be dict-insertion or JAX pytree (sorted keys).
    const bool sorted = (in_sizes[0] != T_STEPS * NUM_INS);
    if (!sorted) {
        int i = 0;
        x    = (const float*)d_in[i++];
        in_W = (const float*)d_in[i++];
        in_b = (const float*)d_in[i++];
        for (int l = 0; l < 2; ++l) {
            Wih[l] = (const float*)d_in[i++];
            Whh[l] = (const float*)d_in[i++];
            bih[l] = (const float*)d_in[i++];
            bhh[l] = (const float*)d_in[i++];
        }
        out_W = (const float*)d_in[i++];
        out_b = (const float*)d_in[i++];
        for (int k = 0; k < 5; ++k) fW[k] = (const float*)d_in[i++];
        for (int k = 0; k < 5; ++k) fb[k] = (const float*)d_in[i++];
    } else {
        int i = 0;
        for (int k = 0; k < 5; ++k) fW[k] = (const float*)d_in[i++];
        for (int k = 0; k < 5; ++k) fb[k] = (const float*)d_in[i++];
        in_W = (const float*)d_in[i++];
        in_b = (const float*)d_in[i++];
        for (int l = 0; l < 2; ++l) {
            Whh[l] = (const float*)d_in[i++];   // sorted: 'Whh' < 'Wih'
            Wih[l] = (const float*)d_in[i++];
            bhh[l] = (const float*)d_in[i++];   // sorted: 'bhh' < 'bih'
            bih[l] = (const float*)d_in[i++];
        }
        out_W = (const float*)d_in[i++];
        out_b = (const float*)d_in[i++];
        x     = (const float*)d_in[i++];
    }

    // Workspace bump allocator (256B aligned)
    char*  ws  = (char*)d_ws;
    size_t off = 0;
    auto alloc = [&](size_t bytes) -> void* {
        void* p = ws + off;
        off = (off + bytes + 255) & ~(size_t)255;
        return p;
    };

    const int XN = T_STEPS * NUM_INS;            // 262144
    __bf16*   xbf  = (__bf16*)alloc((size_t)XN * 2);
    __bf16*   inWT = (__bf16*)alloc((size_t)256 * 256 * 2);
    __bf16*   W1T  = (__bf16*)alloc((size_t)G4 * 256 * 2);
    __bf16*   W2T  = (__bf16*)alloc((size_t)G4 * 256 * 2);
    __bf16*   oWT  = (__bf16*)alloc((size_t)SLOW_PAD * 256 * 2);
    uint32_t* Whp1 = (uint32_t*)alloc((size_t)(HID / 2) * G4 * 4);
    uint32_t* Whp2 = (uint32_t*)alloc((size_t)(HID / 2) * G4 * 4);
    float*    h0   = (float*)alloc((size_t)XN * 4);
    __bf16*   h0b  = (__bf16*)alloc((size_t)XN * 2);
    float*    Abuf = (float*)alloc((size_t)T_STEPS * G4 * 4);  // reused both layers
    float*    h1   = (float*)alloc((size_t)XN * 4);
    __bf16*   h1b  = (__bf16*)alloc((size_t)XN * 2);
    float*    h2   = (float*)alloc((size_t)XN * 4);
    __bf16*   h2b  = (__bf16*)alloc((size_t)XN * 2);
    float*    ubuf = (float*)alloc((size_t)T_STEPS * SLOW_PAD * 4);
    float*    hmid = (float*)alloc((size_t)T_STEPS * 128 * 4);
    int*      flg  = (int*)alloc((size_t)T_STEPS * 4);
    (void)ws_size; (void)out_size;

    auto cdiv = [](int a, int b) { return (a + b - 1) / b; };
    auto gemm_grid = [](int M, int Npad) { return ((Npad / 16) * (M / 16) + 3) / 4; };

    // --- weight prep ---
    k_f32_to_bf16<<<cdiv(XN, 256), 256, 0, stream>>>(x, xbf, XN);
    k_transpose_bf16<<<cdiv(256 * 256, 256), 256, 0, stream>>>(in_W, inWT, 256, 256, 256);
    k_transpose_bf16<<<cdiv(G4 * 256, 256), 256, 0, stream>>>(Wih[0], W1T, 256, G4, G4);
    k_transpose_bf16<<<cdiv(G4 * 256, 256), 256, 0, stream>>>(Wih[1], W2T, 256, G4, G4);
    k_transpose_bf16<<<cdiv(SLOW_PAD * 256, 256), 256, 0, stream>>>(out_W, oWT, 256, SLOW_OUTS, SLOW_PAD);
    k_pack_bf16x2<<<cdiv((HID / 2) * G4, 256), 256, 0, stream>>>(Whh[0], Whp1, HID, G4);
    k_pack_bf16x2<<<cdiv((HID / 2) * G4, 256), 256, 0, stream>>>(Whh[1], Whp2, HID, G4);
    k_zero_f32<<<cdiv(T_STEPS, 256), 256, 0, stream>>>((float*)flg, T_STEPS);

    // --- h0 = relu(x @ in_W + in_b) ---
    k_gemm_bf16<<<gemm_grid(T_STEPS, 256), 128, 0, stream>>>(
        xbf, inWT, in_b, nullptr, h0, 256, h0b, T_STEPS, 256, 256, 1);

    // --- LSTM layer 1 ---
    k_gemm_bf16<<<gemm_grid(T_STEPS, G4), 128, 0, stream>>>(
        h0b, W1T, bih[0], bhh[0], Abuf, G4, nullptr, T_STEPS, G4, G4, 0);
    k_lstm_scan<<<1, 1024, 0, stream>>>(Abuf, Whp1, h1, h1b);

    // --- LSTM layer 2 ---
    k_gemm_bf16<<<gemm_grid(T_STEPS, G4), 128, 0, stream>>>(
        h1b, W2T, bih[1], bhh[1], Abuf, G4, nullptr, T_STEPS, G4, G4, 0);
    k_lstm_scan<<<1, 1024, 0, stream>>>(Abuf, Whp2, h2, h2b);

    // --- u = h2 @ out_W + out_b ---
    k_gemm_bf16<<<gemm_grid(T_STEPS, SLOW_PAD), 128, 0, stream>>>(
        h2b, oWT, out_b, nullptr, ubuf, SLOW_PAD, nullptr, T_STEPS, SLOW_OUTS, SLOW_PAD, 0);

    // --- pipelined fast-weight scan (2 blocks, LDS-resident dflat halves) ---
    FastParams fp;
    for (int k = 0; k < 5; ++k) { fp.W[k] = fW[k]; fp.b[k] = fb[k]; }
    k_fast_pipe<<<2, 1024, (size_t)P_STAGE_B * sizeof(float), stream>>>(
        ubuf, x, fp, hmid, flg, (float*)d_out);
}
